// MultiHeadSelfAttention_1614907704014
// MI455X (gfx1250) — matrix operand order repaired and checked by reading, not compiled, not run on hardware
//
#include <hip/hip_runtime.h>

// ---------------------------------------------------------------------------
// GQA causal self-attention forward for MI455X (gfx1250, wave32, WMMA).
// bf16 inputs / f32 WMMA accumulation, flash-style fused attention.
// Round 2: batched operand loads (load-all-then-wmma) so the compiler can
// clause the global_load_b128s and overlap them with the matrix pipe instead
// of the load->wait 0->wmma serial chain seen in round 1's asm; V fragments
// are fetched early and their latency hidden under the softmax VALU work.
// ---------------------------------------------------------------------------

typedef __attribute__((ext_vector_type(16))) __bf16       bf16x16;
typedef __attribute__((ext_vector_type(8)))  float        f32x8;
typedef __attribute__((ext_vector_type(4)))  unsigned int u32x4;

#define S_LEN 2048
#define NH    16
#define NKV   4
#define HD    64
#define EMB   1024
#define NG    2
#define NROW  4096   // S * G

__device__ __forceinline__ __bf16 f2bf(float f) {
  unsigned u = __builtin_bit_cast(unsigned, f);
  unsigned r = u + 0x7FFFu + ((u >> 16) & 1u);           // round-to-nearest-even
  unsigned short h = (unsigned short)(r >> 16);
  return __builtin_bit_cast(__bf16, h);
}

union Frag { bf16x16 v; u32x4 q[2]; };

// B operand: 16 contiguous bf16 (32 bytes) of one column's k-run.
__device__ __forceinline__ bf16x16 load_b32b(const __bf16* p) {
  Frag f;
  f.q[0] = *(const u32x4*)(p);
  f.q[1] = *(const u32x4*)(p + 8);
  return f.v;
}

// A operand: elems 0..15 = row[base..base+15].
__device__ __forceinline__ bf16x16 load_afrag(const __bf16* row, int base) {
  Frag f;
  f.q[0] = *(const u32x4*)(row + base);
  f.q[1] = *(const u32x4*)(row + base + 8);
  return f.v;
}

__device__ __forceinline__ f32x8 wmma_bf16(bf16x16 a, bf16x16 b, f32x8 c) {
  return __builtin_amdgcn_wmma_f32_16x16x32_bf16(false, a, false, b,
                                                 (short)0, c, false, false);
}

// ---------------------------------------------------------------------------
// fp32 -> bf16 converters (straight and transposed: in[K][N] -> out[N][K])
// ---------------------------------------------------------------------------
__global__ __launch_bounds__(256) void k_cvt(const float* __restrict__ in,
                                             __bf16* __restrict__ out, int n) {
  int i = blockIdx.x * 256 + threadIdx.x;
  if (i < n) out[i] = f2bf(in[i]);
}

__global__ __launch_bounds__(256) void k_cvt_T(const float* __restrict__ in,
                                               __bf16* __restrict__ out,
                                               int K, int N) {
  int i = blockIdx.x * 256 + threadIdx.x;
  if (i < K * N) {
    int k = i / N, n = i - k * N;
    out[(size_t)n * K + k] = f2bf(in[i]);
  }
}

// ---------------------------------------------------------------------------
// WMMA GEMM: C[M,N] = A[M,K](bf16,row-major) x Bt[N,K](bf16, B pre-transposed)
// 8 waves/WG, each wave 16Mx64N (4 accumulators), block tile 128x64.
// k-step 64: 20 b128 loads batched, then 8 interleaved wmmas.
// MODE 0: scatter Q -> [g,h,s,d]    MODE 1: scatter K -> [g,kv,s,d]
// MODE 2: scatter V -> [g,kv,d,s]   MODE 3: fp32 row-major store
// ---------------------------------------------------------------------------
template <int MODE>
__global__ __launch_bounds__(256) void gemm_bf16(const __bf16* __restrict__ A,
                                                 const __bf16* __restrict__ Bt,
                                                 void* __restrict__ Cout,
                                                 int M, int N, int K) {
  (void)M;
  const int lane  = threadIdx.x & 31;
  const int w     = threadIdx.x >> 5;
  const int lhalf = lane >> 4;     // 0/1
  const int lcol  = lane & 15;
  const int m0 = blockIdx.y * 128 + w * 16;
  const int n0 = blockIdx.x * 64;

  const __bf16* arow = A + (size_t)(m0 + lcol) * K;
  f32x8 acc[4] = {};

#pragma unroll 2
  for (int k0 = 0; k0 < K; k0 += 64) {
    if (k0 + 256 < K)
      __builtin_prefetch(arow + k0 + 256, 0, 0);

    // batch all operand loads for two 32-wide k-chunks
    bf16x16 a0 = load_afrag(arow, k0 + (lhalf << 4));
    bf16x16 a1 = load_afrag(arow, k0 + 32 + (lhalf << 4));
    bf16x16 b0[4], b1[4];
#pragma unroll
    for (int t = 0; t < 4; ++t) {
      const __bf16* brow =
          Bt + (size_t)(n0 + t * 16 + lcol) * K + k0 + (lhalf << 4);
      b0[t] = load_b32b(brow);
      b1[t] = load_b32b(brow + 32);
    }
    // 4 independent accumulator chains, interleaved
#pragma unroll
    for (int t = 0; t < 4; ++t) acc[t] = wmma_bf16(a0, b0[t], acc[t]);
#pragma unroll
    for (int t = 0; t < 4; ++t) acc[t] = wmma_bf16(a1, b1[t], acc[t]);
  }

#pragma unroll
  for (int t = 0; t < 4; ++t) {
#pragma unroll
    for (int r = 0; r < 8; ++r) {
      const int mm = m0 + lhalf * 8 + r;
      const int nn = n0 + t * 16 + lcol;
      const float v = acc[t][r];
      if (MODE == 3) {
        ((float*)Cout)[(size_t)mm * N + nn] = v;
      } else {
        const int g = mm & 1, s = mm >> 1;
        __bf16* out = (__bf16*)Cout;
        const int hh = nn >> 6, d = nn & 63;
        if (MODE == 0)
          out[((size_t)(g * NH + hh) * S_LEN + s) * HD + d] = f2bf(v);
        else if (MODE == 1)
          out[((size_t)(g * NKV + hh) * S_LEN + s) * HD + d] = f2bf(v);
        else
          out[((size_t)(g * NKV + hh) * HD + d) * S_LEN + s] = f2bf(v);
      }
    }
  }
}

// ---------------------------------------------------------------------------
// Fused causal flash attention. grid = (S/128, G*H), 8 waves/WG,
// wave owns 16 query rows; K/V operands stream from global in WMMA layout;
// LDS only for per-wave P-tile transpose (C-layout -> A-layout).
// ---------------------------------------------------------------------------
__global__ __launch_bounds__(256) void flash_fwd(const __bf16* __restrict__ Q,
                                                 const __bf16* __restrict__ Kb,
                                                 const __bf16* __restrict__ Vt,
                                                 __bf16* __restrict__ attn) {
  __shared__ __attribute__((aligned(16))) __bf16 pbuf[8][16 * 80];

  const int lane  = threadIdx.x & 31;
  const int w     = threadIdx.x >> 5;
  const int lhalf = lane >> 4;
  const int lcol  = lane & 15;
  const int qt = blockIdx.x;                 // query tile (128 rows)
  const int gh = blockIdx.y;                 // g*16 + h
  const int g = gh >> 4, h = gh & 15, kvh = h >> 2;  // ratio = 4

  const __bf16* Qh = Q  + (size_t)(g * NH  + h)   * S_LEN * HD;
  const __bf16* Kh = Kb + (size_t)(g * NKV + kvh) * S_LEN * HD;
  const __bf16* Vh = Vt + (size_t)(g * NKV + kvh) * HD * S_LEN;

  const int q0 = qt * 128 + w * 16;

  // Q fragments for the two 32-wide k-chunks of HEAD_DIM=64, kept resident.
  const __bf16* qrow = Qh + (size_t)(q0 + lcol) * HD;
  bf16x16 aq0 = load_afrag(qrow, 0  + (lhalf << 4));
  bf16x16 aq1 = load_afrag(qrow, 32 + (lhalf << 4));

  f32x8 o[4] = {};
  float m_i[8], l_i[8];
#pragma unroll
  for (int r = 0; r < 8; ++r) { m_i[r] = -1e30f; l_i[r] = 0.0f; }

  const int nkb = (q0 + 16 + 63) >> 6;       // causal: key blocks up to diag
  for (int kb = 0; kb < nkb; ++kb) {
    const int kbase = kb * 64;
    if (kb + 1 < nkb)
      __builtin_prefetch(Kh + (size_t)(kbase + 64 + lane) * HD, 0, 0);

    // ---- batch all 8 K B-fragments, then 8 interleaved WMMAs (S = Q K^T)
    bf16x16 bk0[4], bk1[4];
#pragma unroll
    for (int nt = 0; nt < 4; ++nt) {
      const __bf16* krow =
          Kh + (size_t)(kbase + nt * 16 + lcol) * HD + (lhalf << 4);
      bk0[nt] = load_b32b(krow);
      bk1[nt] = load_b32b(krow + 32);
    }
    f32x8 c[4];
#pragma unroll
    for (int nt = 0; nt < 4; ++nt) {
      f32x8 z = {};
      c[nt] = wmma_bf16(aq0, bk0[nt], z);
    }
#pragma unroll
    for (int nt = 0; nt < 4; ++nt) c[nt] = wmma_bf16(aq1, bk1[nt], c[nt]);

    // ---- issue V fragment loads NOW; latency hidden under softmax VALU work
    bf16x16 bv0[4], bv1[4];
#pragma unroll
    for (int t = 0; t < 4; ++t) {
      const __bf16* vrow =
          Vh + (size_t)(t * 16 + lcol) * S_LEN + kbase + (lhalf << 4);
      bv0[t] = load_b32b(vrow);
      bv1[t] = load_b32b(vrow + 32);
    }

    // ---- scale by 1/sqrt(64), causal mask
    float s[4][8];
#pragma unroll
    for (int nt = 0; nt < 4; ++nt)
#pragma unroll
      for (int r = 0; r < 8; ++r) {
        const int key = kbase + nt * 16 + lcol;
        const int qi  = q0 + lhalf * 8 + r;
        const float v = c[nt][r] * 0.125f;
        s[nt][r] = (key > qi) ? -1e30f : v;
      }

    // ---- online softmax update (row stats across 16-lane groups)
#pragma unroll
    for (int r = 0; r < 8; ++r) {
      float mx = fmaxf(fmaxf(s[0][r], s[1][r]), fmaxf(s[2][r], s[3][r]));
      mx = fmaxf(mx, __shfl_xor(mx, 1, 32));
      mx = fmaxf(mx, __shfl_xor(mx, 2, 32));
      mx = fmaxf(mx, __shfl_xor(mx, 4, 32));
      mx = fmaxf(mx, __shfl_xor(mx, 8, 32));
      const float mnew  = fmaxf(m_i[r], mx);
      const float alpha = __expf(m_i[r] - mnew);
      m_i[r] = mnew;
      float sum = 0.0f;
#pragma unroll
      for (int nt = 0; nt < 4; ++nt) {
        const float p = __expf(s[nt][r] - mnew);
        s[nt][r] = p;
        sum += p;
      }
      sum += __shfl_xor(sum, 1, 32);
      sum += __shfl_xor(sum, 2, 32);
      sum += __shfl_xor(sum, 4, 32);
      sum += __shfl_xor(sum, 8, 32);
      l_i[r] = l_i[r] * alpha + sum;
#pragma unroll
      for (int t = 0; t < 4; ++t) o[t][r] *= alpha;
    }

    // ---- P tile: C-layout regs -> LDS row-major [16][80] -> A-layout frags
    __bf16* pw = &pbuf[w][0];
#pragma unroll
    for (int nt = 0; nt < 4; ++nt)
#pragma unroll
      for (int r = 0; r < 8; ++r)
        pw[(lhalf * 8 + r) * 80 + nt * 16 + lcol] = f2bf(s[nt][r]);
    asm volatile("s_wait_dscnt 0" ::: "memory");

    bf16x16 ap0, ap1;
    {
      const __bf16* prow = pw + lcol * 80;
      Frag f0, f1;
      f0.q[0] = *(const u32x4*)(prow + 0  + (lhalf << 4));
      f0.q[1] = *(const u32x4*)(prow + 8  + (lhalf << 4));
      f1.q[0] = *(const u32x4*)(prow + 32 + (lhalf << 4));
      f1.q[1] = *(const u32x4*)(prow + 40 + (lhalf << 4));
      ap0 = f0.v; ap1 = f1.v;
    }

    // ---- O += P V : operands already resident, back-to-back WMMAs
#pragma unroll
    for (int t = 0; t < 4; ++t) o[t] = wmma_bf16(ap0, bv0[t], o[t]);
#pragma unroll
    for (int t = 0; t < 4; ++t) o[t] = wmma_bf16(ap1, bv1[t], o[t]);
  }

  // normalize and scatter to attn[(s*G+g)*E + h*64 + d] (bf16, GEMM-ready)
#pragma unroll
  for (int r = 0; r < 8; ++r) {
    const float inv = 1.0f / l_i[r];
    const int   qi  = q0 + lhalf * 8 + r;
#pragma unroll
    for (int t = 0; t < 4; ++t) {
      const int d = t * 16 + lcol;
      attn[((size_t)qi * NG + g) * EMB + h * HD + d] = f2bf(o[t][r] * inv);
    }
  }
}

// ---------------------------------------------------------------------------
// Launch
// ---------------------------------------------------------------------------
extern "C" void kernel_launch(void* const* d_in, const int* in_sizes, int n_in,
                              void* d_out, int out_size, void* d_ws,
                              size_t ws_size, hipStream_t stream) {
  (void)in_sizes; (void)n_in; (void)out_size; (void)ws_size;
  const float* x  = (const float*)d_in[0];
  const float* Wq = (const float*)d_in[1];
  const float* Wk = (const float*)d_in[2];
  const float* Wv = (const float*)d_in[3];
  const float* Wo = (const float*)d_in[4];

  char* p = (char*)d_ws;
  __bf16* Xb  = (__bf16*)p; p += (size_t)NROW * EMB * 2;             // 8 MB
  __bf16* WqT = (__bf16*)p; p += (size_t)EMB * EMB * 2;              // 2 MB
  __bf16* WkT = (__bf16*)p; p += (size_t)256 * EMB * 2;              // .5 MB
  __bf16* WvT = (__bf16*)p; p += (size_t)256 * EMB * 2;              // .5 MB
  __bf16* WoT = (__bf16*)p; p += (size_t)EMB * EMB * 2;              // 2 MB
  __bf16* Qb  = (__bf16*)p; p += (size_t)NG * NH  * S_LEN * HD * 2;  // 8 MB
  __bf16* Kb  = (__bf16*)p; p += (size_t)NG * NKV * S_LEN * HD * 2;  // 2 MB
  __bf16* Vtb = (__bf16*)p; p += (size_t)NG * NKV * S_LEN * HD * 2;  // 2 MB
  __bf16* At  = (__bf16*)p; p += (size_t)NROW * EMB * 2;             // 8 MB

  const int nx = NROW * EMB;
  k_cvt<<<(nx + 255) / 256, 256, 0, stream>>>(x, Xb, nx);
  k_cvt_T<<<(EMB * EMB + 255) / 256, 256, 0, stream>>>(Wq, WqT, EMB, EMB);
  k_cvt_T<<<(EMB * 256 + 255) / 256, 256, 0, stream>>>(Wk, WkT, EMB, 256);
  k_cvt_T<<<(EMB * 256 + 255) / 256, 256, 0, stream>>>(Wv, WvT, EMB, 256);
  k_cvt_T<<<(EMB * EMB + 255) / 256, 256, 0, stream>>>(Wo, WoT, EMB, EMB);

  gemm_bf16<0><<<dim3(EMB / 64, NROW / 128), 256, 0, stream>>>(
      Xb, WqT, Qb, NROW, EMB, EMB);
  gemm_bf16<1><<<dim3(256 / 64, NROW / 128), 256, 0, stream>>>(
      Xb, WkT, Kb, NROW, 256, EMB);
  gemm_bf16<2><<<dim3(256 / 64, NROW / 128), 256, 0, stream>>>(
      Xb, WvT, Vtb, NROW, 256, EMB);

  flash_fwd<<<dim3(S_LEN / 128, NG * NH), 256, 0, stream>>>(Qb, Kb, Vtb, At);

  gemm_bf16<3><<<dim3(EMB / 64, NROW / 128), 256, 0, stream>>>(
      At, WoT, d_out, NROW, EMB, EMB);
}
